// AttentionSingleCritic_75350906241225
// MI455X (gfx1250) — compile-verified, matrix operand order checked by reading, not run on hardware
//
#include <hip/hip_runtime.h>
#include <hip/hip_bf16.h>

typedef __attribute__((ext_vector_type(16))) _Float16 v16h;
typedef __attribute__((ext_vector_type(8)))  _Float16 v8h;
typedef __attribute__((ext_vector_type(8)))  float    v8f;
typedef __attribute__((ext_vector_type(4)))  float    v4f;

#define A_DIM  5
#define B_N    32768
#define SDIM_  48
#define ADIM_  8
#define HID_   256
#define HEADS_ 4
#define DHEAD_ 64
#define LRELU_ 0.01f
#define BN_EPS_ 1e-5f
#define NQ (B_N * ADIM_)   // offset of reg scalar in d_out

// leaky(x) = max(x, a*x) for 0 < a < 1  -> v_mul + v_max (no cndmask)
__device__ __forceinline__ float leaky(float v) { return fmaxf(v, LRELU_ * v); }

// ---------------- WMMA fragment helpers (ISA 7.12.2 layouts) ----------------
// A (16x32 f16): lane m = lane&15, half = lane>>4 selects K block of 8.
// element e -> k = (e>>3)*16 + half*8 + (e&7): two contiguous 8-half runs,
// so each lane's fragment = two aligned 16-byte loads.
__device__ __forceinline__ v16h combine8(v8h lo, v8h hi) {
  return __builtin_shufflevector(lo, hi, 0, 1, 2, 3, 4, 5, 6, 7,
                                 8, 9, 10, 11, 12, 13, 14, 15);
}

__device__ __forceinline__ v16h load_frag_a(const _Float16* base, int ld) {
  const int lane = threadIdx.x & 31;
  const _Float16* p = base + (size_t)(lane & 15) * ld + (((lane >> 4) & 1) << 3);
  const v8h lo = *(const v8h*)p;        // k = hi*8 + 0..7
  const v8h hi = *(const v8h*)(p + 16); // k = 16 + hi*8 + 0..7
  return combine8(lo, hi);
}

// B fragment from swizzled weight storage: per 32x16 tile, [lane][16 halves]
// contiguous -> two 16-byte loads at lane*32 bytes, fully coalesced per wave.
__device__ __forceinline__ v16h load_frag_b_swz(const _Float16* tile) {
  const int lane = threadIdx.x & 31;
  const _Float16* p = tile + lane * 16;
  const v8h lo = *(const v8h*)p;
  const v8h hi = *(const v8h*)(p + 8);
  return combine8(lo, hi);
}

__device__ __forceinline__ v8f wmma_f16(v16h a, v16h b, v8f c) {
  return __builtin_amdgcn_wmma_f32_16x16x32_f16(false, a, false, b, (short)0, c,
                                                false, false);
}

// Swizzled-layout index decode: idx -> (k, n) of a [K][N] matrix.
// idx = ((kt*NT + nt)*32 + lane)*16 + e
__device__ __forceinline__ void swz_decode(int idx, int NT, int& k, int& n) {
  const int e    = idx & 15;
  const int lane = (idx >> 4) & 31;
  const int tile = idx >> 9;
  const int nt   = tile % NT;
  const int kt   = tile / NT;
  k = kt * 32 + ((e >> 3) << 4) + (((lane >> 4) & 1) << 3) + (e & 7);
  n = nt * 16 + (lane & 15);
}

// C/D (16x16 f32): element r -> row = (lane>>4)*8 + r, col = lane&15.

// ---------- Kernel 0: weight conversion to swizzled f16 + reg zero ----------
__global__ void prep_kernel(const float* __restrict__ W_enc,
                            const float* __restrict__ W_key,
                            const float* __restrict__ W_sel,
                            const float* __restrict__ W_c1,
                            const float* __restrict__ W_c2,
                            _Float16* __restrict__ W_ench,   // K 48->64, N 256
                            _Float16* __restrict__ W_keyh,   // K 256, N 256
                            _Float16* __restrict__ W_selh,   // K 256, N 256
                            _Float16* __restrict__ W_c1h,    // K 256, N 256
                            _Float16* __restrict__ W_c2h,    // K 256, N 8->16
                            float* __restrict__ reg_out) {
  const int tid = blockIdx.x * blockDim.x + threadIdx.x;
  if (tid == 0) *reg_out = 0.0f;
  const int n0 = 2 * 16 * 512;   // W_ench  (2 ksteps x 16 ntiles)
  const int n1 = 8 * 16 * 512;   // W_selh
  const int n2 = 8 * 16 * 512;   // W_keyh
  const int n3 = 8 * 16 * 512;   // W_c1h
  const int n4 = 8 * 1 * 512;    // W_c2h   (8 ksteps x 1 ntile)
  const int total = n0 + n1 + n2 + n3 + n4;
  for (int idx = tid; idx < total; idx += gridDim.x * blockDim.x) {
    int i = idx, k, n;
    if (i < n0) {
      swz_decode(i, 16, k, n);
      W_ench[i] = (_Float16)((k < SDIM_) ? W_enc[k * HID_ + n] : 0.0f);
    } else if ((i -= n0) < n1) {
      swz_decode(i, 16, k, n);
      W_selh[i] = (_Float16)W_sel[((size_t)(n >> 6) * HID_ + k) * DHEAD_ + (n & 63)];
    } else if ((i -= n1) < n2) {
      swz_decode(i, 16, k, n);
      W_keyh[i] = (_Float16)W_key[((size_t)(n >> 6) * HID_ + k) * DHEAD_ + (n & 63)];
    } else if ((i -= n2) < n3) {
      swz_decode(i, 16, k, n);
      W_c1h[i] = (_Float16)W_c1[k * HID_ + n];
    } else {
      i -= n3;
      swz_decode(i, 1, k, n);
      W_c2h[i] = (_Float16)((n < ADIM_) ? W_c2[k * ADIM_ + n] : 0.0f);
    }
  }
}

// ---------------- Kernel 1: BN mean / rstd per (a, s) channel ----------------
__global__ void bn_stats_kernel(const float* __restrict__ states,
                                float* __restrict__ mu, float* __restrict__ rstd) {
  const int ch = blockIdx.x;               // 0 .. A_DIM*SDIM_-1
  const int a = ch / SDIM_, s = ch % SDIM_;
  const float* p = states + (size_t)a * B_N * SDIM_ + s;
  float s1 = 0.f, s2 = 0.f;
  for (int b = threadIdx.x; b < B_N; b += blockDim.x) {
    const float v = p[(size_t)b * SDIM_];
    s1 += v; s2 += v * v;
  }
  __shared__ float r1[256], r2[256];
  r1[threadIdx.x] = s1; r2[threadIdx.x] = s2;
  __syncthreads();
  for (int off = 128; off > 0; off >>= 1) {
    if ((int)threadIdx.x < off) {
      r1[threadIdx.x] += r1[threadIdx.x + off];
      r2[threadIdx.x] += r2[threadIdx.x + off];
    }
    __syncthreads();
  }
  if (threadIdx.x == 0) {
    const float m = r1[0] * (1.0f / B_N);
    const float v = r2[0] * (1.0f / B_N) - m * m;
    mu[ch] = m;
    rstd[ch] = rsqrtf(v + BN_EPS_);
  }
}

// Shared helper: stage normalized xn tile (128 rows x 64 padded K) into LDS.
__device__ __forceinline__ void stage_xn(const float* __restrict__ states,
                                         const float* __restrict__ mu,
                                         const float* __restrict__ rstd,
                                         int a, int row0,
                                         _Float16 (*xs)[64]) {
  for (int idx = threadIdx.x; idx < 128 * 64; idx += blockDim.x) {
    const int r = idx >> 6, c = idx & 63;
    float v = 0.f;
    if (c < SDIM_)
      v = (states[((size_t)a * B_N + row0 + r) * SDIM_ + c] - mu[a * SDIM_ + c])
          * rstd[a * SDIM_ + c];
    xs[r][c] = (_Float16)v;
  }
}

// Shared helper: one wave's 16x256 enc tile into its LDS stage buffer.
__device__ __forceinline__ void enc_tile_to_stage(const _Float16 (*xs)[64],
                                                  const _Float16* __restrict__ W_ench,
                                                  const float* __restrict__ b_enc,
                                                  _Float16* __restrict__ stage_w) {
  const int wid = threadIdx.x >> 5;
  const int lane = threadIdx.x & 31;
  const int col16 = lane & 15;
  const int rbase = (lane >> 4) << 3;
  const v16h a0 = load_frag_a(&xs[wid * 16][0], 64);
  const v16h a1 = load_frag_a(&xs[wid * 16][32], 64);
#pragma unroll
  for (int nt = 0; nt < 16; ++nt) {
    v8f acc = {};
    acc = wmma_f16(a0, load_frag_b_swz(W_ench + (0 * 16 + nt) * 512), acc);
    acc = wmma_f16(a1, load_frag_b_swz(W_ench + (1 * 16 + nt) * 512), acc);
    const int col = nt * 16 + col16;
#pragma unroll
    for (int r = 0; r < 8; ++r)
      stage_w[(rbase + r) * HID_ + col] = (_Float16)leaky(acc[r] + b_enc[col]);
  }
}

// ------- Kernel 2: enc[0] = leaky(xn @ W_enc + b), materialized to HBM -------
// grid = B_N/128, 256 threads (8 waves). Wave = 16 rows x 256 cols.
__global__ void enc0_kernel(const float* __restrict__ states,
                            const float* __restrict__ mu,
                            const float* __restrict__ rstd,
                            const _Float16* __restrict__ W_ench,
                            const float* __restrict__ b_enc,
                            _Float16* __restrict__ enc_h) {
  __shared__ _Float16 xs[128][64];                         // 16 KB
  __shared__ alignas(16) _Float16 stage[8][16 * HID_];     // 64 KB
  const int row0 = blockIdx.x * 128;
  stage_xn(states, mu, rstd, 0, row0, xs);
  __syncthreads();

  const int wid = threadIdx.x >> 5;
  const int lane = threadIdx.x & 31;
  enc_tile_to_stage(xs, W_ench, b_enc, stage[wid]);

  // wave-private region: DS ordering within the wave suffices.
  // 16x256 f16 strip is 8 KB contiguous in enc_h.
  _Float16* dst = enc_h + (size_t)(row0 + wid * 16) * HID_;
#pragma unroll
  for (int it = 0; it < 16; ++it) {
    const int chunk = it * 32 + lane;
    *(v8h*)(dst + chunk * 8) = *(const v8h*)(&stage[wid][chunk * 8]);
  }
}

// ---- Kernel 3: dual GEMM sharing A: sel = enc0@Wsel, h = leaky(enc0@Wc1+b) ----
__global__ void gemm_dual_kernel(const _Float16* __restrict__ Amat,
                                 const _Float16* __restrict__ B1,   // swizzled
                                 _Float16* __restrict__ Out1,
                                 const _Float16* __restrict__ B2,   // swizzled
                                 const float* __restrict__ bias2,
                                 _Float16* __restrict__ Out2) {
  __shared__ alignas(16) _Float16 stage[8][16 * HID_];     // 64 KB
  const int wid = threadIdx.x >> 5;
  const int lane = threadIdx.x & 31;
  const int wrow = blockIdx.x * 128 + wid * 16;
  const int col16 = lane & 15;
  const int rbase = (lane >> 4) << 3;
  v16h afr[8];
#pragma unroll
  for (int ks = 0; ks < 8; ++ks)
    afr[ks] = load_frag_a(Amat + (size_t)wrow * HID_ + ks * 32, HID_);

  for (int m = 0; m < 2; ++m) {
    const _Float16* Bmat = m ? B2 : B1;
#pragma unroll
    for (int nt = 0; nt < 16; ++nt) {
      v8f acc = {};
#pragma unroll
      for (int ks = 0; ks < 8; ++ks)
        acc = wmma_f16(afr[ks], load_frag_b_swz(Bmat + (ks * 16 + nt) * 512), acc);
      const int col = nt * 16 + col16;
#pragma unroll
      for (int r = 0; r < 8; ++r) {
        float v = acc[r];
        if (m) v = leaky(v + bias2[col]);
        stage[wid][(rbase + r) * HID_ + col] = (_Float16)v;
      }
    }
    _Float16* dst = (m ? Out2 : Out1) + (size_t)wrow * HID_;
#pragma unroll
    for (int it = 0; it < 16; ++it) {
      const int chunk = it * 32 + lane;
      *(v8h*)(dst + chunk * 8) = *(const v8h*)(&stage[wid][chunk * 8]);
    }
  }
}

// ---- Kernel 4: fused enc[a] (a=1..4) + keys + logits -> reg accumulation ----
// enc[1..4] never touches HBM: computed into the wave's LDS stage, re-read as
// WMMA A-fragments for the key GEMMs, contracted against sel in registers.
// grid = (B_N/128, 4), 256 threads. reg = 0.001 * sum_k mean_{b,a}(logits^2).
__global__ void enc_logits_kernel(const float* __restrict__ states,
                                  const float* __restrict__ mu,
                                  const float* __restrict__ rstd,
                                  const _Float16* __restrict__ W_ench,
                                  const float* __restrict__ b_enc,
                                  const _Float16* __restrict__ W_keyh, // swizzled
                                  const _Float16* __restrict__ sel_h,
                                  float* __restrict__ reg_out) {
  __shared__ _Float16 xs[128][64];                         // 16 KB
  __shared__ alignas(16) _Float16 stage[8][16 * HID_];     // 64 KB
  const int a = blockIdx.y + 1;
  const int row0 = blockIdx.x * 128;
  stage_xn(states, mu, rstd, a, row0, xs);
  __syncthreads();

  const int wid = threadIdx.x >> 5;
  const int lane = threadIdx.x & 31;
  const int col16 = lane & 15;
  const int rbase = (lane >> 4) << 3;
  enc_tile_to_stage(xs, W_ench, b_enc, stage[wid]);

  // re-load the freshly computed enc tile as A fragments (wave-private LDS)
  v16h afr[8];
#pragma unroll
  for (int ks = 0; ks < 8; ++ks)
    afr[ks] = load_frag_a(&stage[wid][ks * 32], HID_);

  const int grow0 = row0 + wid * 16;
  float local = 0.f;
#pragma unroll
  for (int k = 0; k < HEADS_; ++k) {
    float p[8];
#pragma unroll
    for (int r = 0; r < 8; ++r) p[r] = 0.f;
#pragma unroll
    for (int dt = 0; dt < 4; ++dt) {
      v8f acc = {};
#pragma unroll
      for (int ks = 0; ks < 8; ++ks)
        acc = wmma_f16(afr[ks],
                       load_frag_b_swz(W_keyh + (ks * 16 + (k * 4 + dt)) * 512),
                       acc);
      const int scol = k * DHEAD_ + dt * 16 + col16;
#pragma unroll
      for (int r = 0; r < 8; ++r)
        p[r] += acc[r] * (float)sel_h[(size_t)(grow0 + rbase + r) * HID_ + scol];
    }
    // reduce dot products across the 16-lane half (cols of the key tile)
#pragma unroll
    for (int m = 8; m >= 1; m >>= 1)
#pragma unroll
      for (int r = 0; r < 8; ++r) p[r] += __shfl_xor(p[r], m, 32);
    if ((lane & 15) == 0) {
#pragma unroll
      for (int r = 0; r < 8; ++r) local += p[r] * p[r];
    }
  }
#pragma unroll
  for (int m = 16; m >= 1; m >>= 1) local += __shfl_xor(local, m, 32);

  __shared__ float bsum;
  if (threadIdx.x == 0) bsum = 0.f;
  __syncthreads();
  if (lane == 0) atomicAdd(&bsum, local);
  __syncthreads();
  if (threadIdx.x == 0) {
    const float SCALE = 0.001f / ((float)B_N * 4.0f);
    atomicAdd(reg_out, bsum * SCALE);
  }
}

// ------- Kernel 5: all_q = h @ W_c2 + b_c2 (N padded 8->16) -------
__global__ void allq_kernel(const _Float16* __restrict__ h_h,
                            const _Float16* __restrict__ W_c2h,   // swizzled
                            const float* __restrict__ b_c2,
                            float* __restrict__ out) {
  __shared__ alignas(16) float stageq[8][16 * ADIM_];     // 4 KB
  const int wid = threadIdx.x >> 5;
  const int lane = threadIdx.x & 31;
  const int wrow = blockIdx.x * 128 + wid * 16;
  v8f acc = {};
#pragma unroll
  for (int ks = 0; ks < 8; ++ks) {
    const v16h afr = load_frag_a(h_h + (size_t)wrow * HID_ + ks * 32, HID_);
    acc = wmma_f16(afr, load_frag_b_swz(W_c2h + ks * 512), acc);
  }
  const int col = lane & 15;
  const int rbase = (lane >> 4) << 3;
  if (col < ADIM_) {
#pragma unroll
    for (int r = 0; r < 8; ++r)
      stageq[wid][(rbase + r) * ADIM_ + col] = acc[r] + b_c2[col];
  }
  // wave-private region; 16x8 f32 block is 512 contiguous bytes in out
  float* dst = out + (size_t)wrow * ADIM_;
  *(v4f*)(dst + lane * 4) = *(const v4f*)(&stageq[wid][lane * 4]);
}

// -------------------------------- launcher ----------------------------------
extern "C" void kernel_launch(void* const* d_in, const int* in_sizes, int n_in,
                              void* d_out, int out_size, void* d_ws, size_t ws_size,
                              hipStream_t stream) {
  (void)in_sizes; (void)n_in; (void)out_size; (void)ws_size;
  const float* states = (const float*)d_in[0];
  const float* W_enc  = (const float*)d_in[1];
  const float* b_enc  = (const float*)d_in[2];
  const float* W_key  = (const float*)d_in[3];
  const float* W_sel  = (const float*)d_in[4];
  // d_in[5] = W_val, d_in[6] = b_val : dead (multiplied by 0.0 in reference)
  const float* W_c1   = (const float*)d_in[7];
  const float* b_c1   = (const float*)d_in[8];
  const float* W_c2   = (const float*)d_in[9];
  const float* b_c2   = (const float*)d_in[10];
  float* out = (float*)d_out;

  char* w = (char*)d_ws;
  auto alloc = [&](size_t bytes) -> char* {
    char* p = w;
    w += (bytes + 255) & ~(size_t)255;
    return p;
  };
  float*    mu     = (float*)alloc(A_DIM * SDIM_ * 4);
  float*    rstd   = (float*)alloc(A_DIM * SDIM_ * 4);
  _Float16* W_ench = (_Float16*)alloc(64 * HID_ * 2);
  _Float16* W_selh = (_Float16*)alloc(HID_ * HID_ * 2);
  _Float16* W_keyh = (_Float16*)alloc(HID_ * HID_ * 2);
  _Float16* W_c1h  = (_Float16*)alloc(HID_ * HID_ * 2);
  _Float16* W_c2h  = (_Float16*)alloc(HID_ * 16 * 2);
  _Float16* enc_h  = (_Float16*)alloc((size_t)B_N * HID_ * 2);   // enc[0] only
  _Float16* sel_h  = (_Float16*)alloc((size_t)B_N * HID_ * 2);
  _Float16* h_h    = (_Float16*)alloc((size_t)B_N * HID_ * 2);

  prep_kernel<<<848, 256, 0, stream>>>(W_enc, W_key, W_sel, W_c1, W_c2,
                                       W_ench, W_keyh, W_selh, W_c1h, W_c2h,
                                       out + NQ);
  bn_stats_kernel<<<A_DIM * SDIM_, 256, 0, stream>>>(states, mu, rstd);
  enc0_kernel<<<B_N / 128, 256, 0, stream>>>(states, mu, rstd, W_ench, b_enc,
                                             enc_h);
  gemm_dual_kernel<<<B_N / 128, 256, 0, stream>>>(enc_h, W_selh, sel_h,
                                                  W_c1h, b_c1, h_h);
  enc_logits_kernel<<<dim3(B_N / 128, 4), 256, 0, stream>>>(states, mu, rstd,
                                                            W_ench, b_enc,
                                                            W_keyh, sel_h,
                                                            out + NQ);
  allq_kernel<<<B_N / 128, 256, 0, stream>>>(h_h, W_c2h, b_c2, out);
}
